// FlaxDbrxBlock_67997922230330
// MI455X (gfx1250) — compile-verified, matrix operand order checked by reading, not run on hardware
//
#include <hip/hip_runtime.h>

// MoE (DBRX-style) for MI455X / gfx1250, wave32 + WMMA bf16 + async LDS copy.
// T=4096 tokens, D=2048, F=4096, E=8 experts, top-2 routing.

#define E_ 8
#define F_ 4096
#define D_ 2048
#define T_ 4096

typedef __bf16 bf16;
typedef bf16 v16bf __attribute__((ext_vector_type(16)));
typedef float v8f   __attribute__((ext_vector_type(8)));
typedef int  v4i    __attribute__((ext_vector_type(4)));

// native fp32 -> bf16 (compiler lowers to v_cvt_*bf16_f32 on gfx1250)
__device__ __forceinline__ bf16 f2bf(float f) { return (bf16)f; }

// gfx1250 async global->LDS copy (ASYNCcnt-tracked), 16 bytes per lane.
// Builtin signature (from hipcc diagnostic): (AS1 v4i*, AS3 v4i*, imm, imm).
__device__ __forceinline__ void async_ld_b128(const bf16* gsrc, bf16* ldst) {
  __builtin_amdgcn_global_load_async_to_lds_b128(
      (__attribute__((address_space(1))) v4i*)gsrc,
      (__attribute__((address_space(3))) v4i*)ldst, 0, 0);
}
__device__ __forceinline__ void wait_async0() {
  asm volatile("s_wait_asynccnt 0x0" ::: "memory");
}

// ---------------- init: zero output accumulator + expert counters -----------
__global__ void moe_init_kernel(float* __restrict__ out, int* __restrict__ counts) {
  size_t i = ((size_t)blockIdx.x * blockDim.x + threadIdx.x) * 8;
  #pragma unroll
  for (int j = 0; j < 8; ++j) out[i + j] = 0.0f;   // grid sized exactly T_*D_/8 threads
  if (blockIdx.x == 0 && threadIdx.x < E_) counts[threadIdx.x] = 0;
}

// ---------------- convert x (fp32) -> bf16 once -----------------------------
__global__ void moe_cvt_kernel(const float* __restrict__ x, bf16* __restrict__ xbf) {
  size_t i = ((size_t)blockIdx.x * blockDim.x + threadIdx.x) * 8;
  #pragma unroll
  for (int j = 0; j < 8; ++j) xbf[i + j] = f2bf(x[i + j]);
}

// ---------------- router: logits -> softmax -> top2 -> L1 norm --------------
__global__ void moe_router_kernel(const float* __restrict__ x,
                                  const float* __restrict__ rw,   // [D, E] row-major
                                  float* __restrict__ coef,       // [E][T]
                                  int* __restrict__ counts,       // [E]
                                  int* __restrict__ tl) {         // [E][T]
  int t = blockIdx.x * blockDim.x + threadIdx.x;                  // 0..T_-1
  const float* xr = x + (size_t)t * D_;
  float lg[E_];
  #pragma unroll
  for (int e = 0; e < E_; ++e) lg[e] = 0.0f;
  for (int d = 0; d < D_; ++d) {
    float xv = xr[d];
    #pragma unroll
    for (int e = 0; e < E_; ++e) lg[e] += xv * rw[d * E_ + e];
  }
  float mx = lg[0];
  #pragma unroll
  for (int e = 1; e < E_; ++e) mx = fmaxf(mx, lg[e]);
  float w[E_], sum = 0.0f;
  #pragma unroll
  for (int e = 0; e < E_; ++e) { w[e] = __expf(lg[e] - mx); sum += w[e]; }
  float inv = 1.0f / sum;
  #pragma unroll
  for (int e = 0; e < E_; ++e) w[e] *= inv;
  // top-2 (first occurrence wins ties, like lax.top_k)
  int e0 = 0; float w0 = w[0];
  #pragma unroll
  for (int e = 1; e < E_; ++e) { if (w[e] > w0) { w0 = w[e]; e0 = e; } }
  int e1 = -1; float w1v = -1.0f;
  #pragma unroll
  for (int e = 0; e < E_; ++e) { if (e != e0 && w[e] > w1v) { w1v = w[e]; e1 = e; } }
  float s = w0 + w1v;
  #pragma unroll
  for (int e = 0; e < E_; ++e) coef[e * T_ + t] = 0.0f;
  coef[e0 * T_ + t] = w0 / s;
  coef[e1 * T_ + t] = w1v / s;
  int p0 = atomicAdd(&counts[e0], 1); tl[e0 * T_ + p0] = t;
  int p1 = atomicAdd(&counts[e1], 1); tl[e1 * T_ + p1] = t;
}

// ---------------- GEMM1: h = silu(x*w1^T) * (x*v1^T)  (gathered rows) -------
// Block: 64 (gathered rows) x 128 (F cols), BK=32, 256 thr = 8 waves.
// Wave w: M-subtile = w>>1, N-subtiles = 4*(w&1) .. +3.
__global__ __launch_bounds__(256) void moe_gemm1_kernel(
    const bf16* __restrict__ xbf, const float* __restrict__ w1g,
    const float* __restrict__ v1g, bf16* __restrict__ h,
    const int* __restrict__ tl, const int* __restrict__ counts, int e) {
  const int tid  = threadIdx.x;
  const int lane = tid & 31;
  const int wave = tid >> 5;
  const int waveM = wave >> 1;
  const int nBase = (wave & 1) * 4;
  const int f0 = blockIdx.x * 128;
  const int r0 = blockIdx.y * 64;
  const int cnt = counts[e];
  if (r0 >= cnt) return;
  const int* tlE = tl + e * T_;

  __shared__ __align__(32) bf16 aS[4][32][16];    // 4 KB
  __shared__ __align__(32) bf16 b1S[8][32][16];   // 8 KB
  __shared__ __align__(32) bf16 b2S[8][32][16];   // 8 KB

  v8f zf = {0.f,0.f,0.f,0.f,0.f,0.f,0.f,0.f};
  v8f acc1[4] = {zf, zf, zf, zf};
  v8f acc2[4] = {zf, zf, zf, zf};

  // A staging: 2048 bf16 / 256 thr = 8 each (contiguous k-chunk of one row)
  const int aEl = tid * 8;
  const int aM  = aEl >> 5;
  const int aK  = aEl & 31;                       // 0,8,16,24
  const int row = r0 + aM;
  const int tok = (row < cnt) ? tlE[row] : tlE[0];
  const bf16* aSrc = xbf + (size_t)tok * D_ + aK;
  bf16* aDst = &aS[aM >> 4][(aM & 15) + (((aK >> 3) & 1) << 4)]
                  [(aK & 7) + ((aK >> 4) << 3)];

  // B staging: 4096 fp32 per matrix / 256 thr = 16 each (one f-row k-chunk)
  const int bEl = tid * 16;
  const int bN  = bEl >> 5;
  const int bK  = bEl & 31;                       // 0 or 16
  const float* b1Base = w1g + ((size_t)e * F_ + f0 + bN) * D_ + bK;
  const float* b2Base = v1g + ((size_t)e * F_ + f0 + bN) * D_ + bK;
  const int bSubN = bN >> 4;
  const int bNN   = bN & 15;

  for (int k0 = 0; k0 < D_; k0 += 32) {
    __syncthreads();
    async_ld_b128(aSrc + k0, aDst);               // x tile: async DMA to LDS
    #pragma unroll
    for (int c = 0; c < 2; ++c) {
      int kc = bK + c * 8;
      int ld = bNN + (((kc >> 3) & 1) << 4);
      int sd = (kc >> 4) << 3;
      __align__(16) bf16 t1[8], t2[8];
      #pragma unroll
      for (int j = 0; j < 8; ++j) { t1[j] = f2bf(b1Base[k0 + c * 8 + j]);
                                    t2[j] = f2bf(b2Base[k0 + c * 8 + j]); }
      *reinterpret_cast<uint4*>(&b1S[bSubN][ld][sd]) = *reinterpret_cast<const uint4*>(t1);
      *reinterpret_cast<uint4*>(&b2S[bSubN][ld][sd]) = *reinterpret_cast<const uint4*>(t2);
    }
    if (k0 + 32 < D_) {
      __builtin_prefetch(b1Base + k0 + 32, 0, 0);   // global_prefetch_b8
      __builtin_prefetch(b2Base + k0 + 32, 0, 0);
    }
    wait_async0();
    __syncthreads();
    // preload all fragments, then back-to-back WMMAs
    v16bf a = *reinterpret_cast<const v16bf*>(&aS[waveM][lane][0]);
    v16bf b1f[4], b2f[4];
    #pragma unroll
    for (int t = 0; t < 4; ++t) {
      b1f[t] = *reinterpret_cast<const v16bf*>(&b1S[nBase + t][lane][0]);
      b2f[t] = *reinterpret_cast<const v16bf*>(&b2S[nBase + t][lane][0]);
    }
    #pragma unroll
    for (int t = 0; t < 4; ++t) {
      acc1[t] = __builtin_amdgcn_wmma_f32_16x16x32_bf16(false, a, false, b1f[t],
                                                        (short)0, acc1[t], false, false);
      acc2[t] = __builtin_amdgcn_wmma_f32_16x16x32_bf16(false, a, false, b2f[t],
                                                        (short)0, acc2[t], false, false);
    }
  }

  // epilogue: GLU, store h in bf16 (C layout: VGPR r -> M = r + 8*(lane>=16))
  const int mBase = waveM * 16 + ((lane >> 4) << 3);
  const int nCol  = lane & 15;
  #pragma unroll
  for (int t = 0; t < 4; ++t) {
    #pragma unroll
    for (int r = 0; r < 8; ++r) {
      float x1 = acc1[t][r];
      float x2 = acc2[t][r];
      float hv = (x1 / (1.0f + __expf(-x1))) * x2;   // silu(x1)*x2
      int m = r0 + mBase + r;
      int f = f0 + (nBase + t) * 16 + nCol;
      h[(size_t)m * F_ + f] = f2bf(hv);
    }
  }
}

// ---------------- GEMM2: out[tok] += coef * (h @ w2[e]) ---------------------
__global__ __launch_bounds__(256) void moe_gemm2_kernel(
    const bf16* __restrict__ h, const float* __restrict__ w2g,
    float* __restrict__ out, const int* __restrict__ tl,
    const int* __restrict__ counts, const float* __restrict__ coef, int e) {
  const int tid  = threadIdx.x;
  const int lane = tid & 31;
  const int wave = tid >> 5;
  const int waveM = wave >> 1;
  const int nBase = (wave & 1) * 4;
  const int d0 = blockIdx.x * 128;
  const int r0 = blockIdx.y * 64;
  const int cnt = counts[e];
  if (r0 >= cnt) return;
  const int* tlE = tl + e * T_;

  __shared__ __align__(32) bf16 aS[4][32][16];
  __shared__ __align__(32) bf16 bS[8][32][16];

  v8f zf = {0.f,0.f,0.f,0.f,0.f,0.f,0.f,0.f};
  v8f acc[4] = {zf, zf, zf, zf};

  const int aEl = tid * 8;
  const int aM  = aEl >> 5;
  const int aK  = aEl & 31;
  const bf16* aSrc = h + (size_t)(r0 + aM) * F_ + aK;
  bf16* aDst = &aS[aM >> 4][(aM & 15) + (((aK >> 3) & 1) << 4)]
                  [(aK & 7) + ((aK >> 4) << 3)];

  // B from w2 (row-major [F,D]): element (k,n) = w2[(e*F + k0+k)*D + d0+n]
  const int bEl   = tid * 16;
  const int bKrow = bEl >> 7;           // 0..31
  const int bN16  = bEl & 127;          // multiple of 16
  const int bSubN = bN16 >> 4;
  const float* bBase = w2g + ((size_t)e * F_ + bKrow) * D_ + d0 + bN16;
  const int bLaneBase = ((bKrow >> 3) & 1) << 4;
  const int bSlot     = (bKrow & 7) + ((bKrow >> 4) << 3);

  for (int k0 = 0; k0 < F_; k0 += 32) {
    __syncthreads();
    async_ld_b128(aSrc + k0, aDst);               // h tile: async DMA to LDS
    const float* bp = bBase + (size_t)k0 * D_;
    #pragma unroll
    for (int j = 0; j < 16; ++j) bS[bSubN][bLaneBase + j][bSlot] = f2bf(bp[j]);
    if (k0 + 32 < F_) __builtin_prefetch(bp + (size_t)32 * D_, 0, 0);
    wait_async0();
    __syncthreads();
    v16bf a = *reinterpret_cast<const v16bf*>(&aS[waveM][lane][0]);
    v16bf bf[4];
    #pragma unroll
    for (int t = 0; t < 4; ++t)
      bf[t] = *reinterpret_cast<const v16bf*>(&bS[nBase + t][lane][0]);
    #pragma unroll
    for (int t = 0; t < 4; ++t)
      acc[t] = __builtin_amdgcn_wmma_f32_16x16x32_bf16(false, a, false, bf[t],
                                                       (short)0, acc[t], false, false);
  }

  const int mBase = waveM * 16 + ((lane >> 4) << 3);
  const int nCol  = lane & 15;
  #pragma unroll
  for (int t = 0; t < 4; ++t) {
    #pragma unroll
    for (int r = 0; r < 8; ++r) {
      int row = r0 + mBase + r;
      if (row < cnt) {
        int tok = tlE[row];
        float c = coef[e * T_ + tok];
        out[(size_t)tok * D_ + d0 + (nBase + t) * 16 + nCol] += c * acc[t][r];
      }
    }
  }
}

// ---------------- launch ----------------------------------------------------
extern "C" void kernel_launch(void* const* d_in, const int* in_sizes, int n_in,
                              void* d_out, int out_size, void* d_ws, size_t ws_size,
                              hipStream_t stream) {
  const float* x  = (const float*)d_in[0];
  const float* rw = (const float*)d_in[1];
  const float* w1 = (const float*)d_in[2];
  const float* v1 = (const float*)d_in[3];
  const float* w2 = (const float*)d_in[4];
  float* out = (float*)d_out;

  char* ws = (char*)d_ws;
  int*   counts = (int*)(ws + 0);                               // 32 B
  int*   tl     = (int*)(ws + 256);                             // E*T*4   = 128 KB
  float* coef   = (float*)(ws + 256 + (size_t)E_ * T_ * 4);     // E*T*4   = 128 KB
  bf16*  xbf    = (bf16*)(ws + 262400);                         // T*D*2   = 16 MB
  bf16*  hbuf   = (bf16*)(ws + 262400 + (size_t)T_ * D_ * 2);   // T*F*2   = 32 MB

  moe_init_kernel<<<(T_ * D_) / (256 * 8), 256, 0, stream>>>(out, counts);
  moe_cvt_kernel <<<(T_ * D_) / (256 * 8), 256, 0, stream>>>(x, xbf);
  moe_router_kernel<<<T_ / 256, 256, 0, stream>>>(x, rw, coef, counts, tl);

  for (int e = 0; e < E_; ++e) {
    moe_gemm1_kernel<<<dim3(F_ / 128, T_ / 64), 256, 0, stream>>>(
        xbf, w1, v1, hbuf, tl, counts, e);
    moe_gemm2_kernel<<<dim3(D_ / 128, T_ / 64), 256, 0, stream>>>(
        hbuf, w2, out, tl, counts, coef, e);
  }
}
